// SelfAttention_10230612099752
// MI455X (gfx1250) — matrix operand order repaired and checked by reading, not compiled, not run on hardware
//
#include <hip/hip_runtime.h>

// ---------------------------------------------------------------------------
// MI455X (gfx1250) self-attention: B=8, S=2048, D=U=1024, fp32 in/out.
// All matmuls on v_wmma_f32_16x16x32_bf16 (wave32). Tiles staged with
// CDNA5 async global->LDS copies (ASYNCcnt) and double buffering.
// Pipeline: cvt(X)->bf16, transpose-cvt(W), QKV GEMMs, scores GEMM,
// row softmax, P@V GEMM -> fp32 out. Intermediates live in d_ws (L2-resident).
// ---------------------------------------------------------------------------

#define BDIM 8
#define SDIM 2048
#define DDIM 1024
#define UDIM 1024
#define MROWS (BDIM * SDIM) /* 16384 */

typedef __attribute__((ext_vector_type(16))) __bf16 v16bf;
typedef __attribute__((ext_vector_type(8)))  __bf16 v8bf;
typedef __attribute__((ext_vector_type(4)))  __bf16 v4bf;
typedef __attribute__((ext_vector_type(8)))  float  v8f;

__device__ __forceinline__ __bf16 f2bf(float f) {
  unsigned u = __builtin_bit_cast(unsigned, f);
  u += 0x7FFFu + ((u >> 16) & 1u);           // round-to-nearest-even
  unsigned short h = (unsigned short)(u >> 16);
  return __builtin_bit_cast(__bf16, h);
}

struct BFPair { v8bf lo, hi; };              // 32B -> bit_cast to v16bf

#define LDS_STRIDE 40                        // 32 halves + 8 pad (80B rows, 16B aligned)

// Issue one 16B async global->LDS copy (ASYNCcnt-tracked, no VGPR data).
__device__ __forceinline__ void async_cp16(const __bf16* gsrc, const __bf16* ldst) {
  unsigned lds = (unsigned)(uintptr_t)ldst;               // low 32b = LDS offset
  unsigned long long ga = (unsigned long long)(uintptr_t)gsrc;
  asm volatile("global_load_async_to_lds_b128 %0, %1, off"
               :: "v"(lds), "v"(ga) : "memory");
}
__device__ __forceinline__ void wait_async0() {
  asm volatile("s_wait_asynccnt 0x0" ::: "memory");
}

// C[256x128] tile = A[256xK] * B^T where Bsrc is row-major [N][K].
// 8 waves in 4x2 grid; each wave owns 64x64 (4x4 WMMA frags, 16 v8f accums).
// EPI: 0 = bf16 row-major, 1 = bf16 transposed (C^T[n][m]), 2 = f32 row-major*scale
template <int EPI>
__launch_bounds__(256)
__global__ void gemm_bf16_wmma(const __bf16* __restrict__ A,
                               const __bf16* __restrict__ B,
                               void* __restrict__ Cout,
                               int ldA, int ldB, int ldC, int Kd,
                               long long offA, long long offB, long long offC,
                               float scale) {
  __shared__ __bf16 As[2][256 * LDS_STRIDE];   // 40 KB
  __shared__ __bf16 Bs[2][128 * LDS_STRIDE];   // 20 KB

  const int bz = blockIdx.z;
  A += (long long)bz * offA;
  B += (long long)bz * offB;

  const int m0   = blockIdx.x * 256;
  const int n0   = blockIdx.y * 128;
  const int tid  = threadIdx.x;
  const int lane = tid & 31;
  const int wave = tid >> 5;
  const int wm   = wave >> 1;   // 0..3 : 64-row slab
  const int wn   = wave & 1;    // 0..1 : 64-col slab
  const int g    = lane >> 4;   // lane group (ISA 16-bit A/B layout)
  const int r    = lane & 15;

  const v8f vzero = {0.f,0.f,0.f,0.f,0.f,0.f,0.f,0.f};
  v8f acc[4][4];
#pragma unroll
  for (int mi = 0; mi < 4; ++mi)
#pragma unroll
    for (int ni = 0; ni < 4; ++ni) acc[mi][ni] = vzero;

  // Async-stage one 256x32 A tile + 128x32 B tile (24 KB): 6x16B per thread.
  auto stage = [&](int buf, int kt) {
#pragma unroll
    for (int j = 0; j < 4; ++j) {                 // A: 1024 chunks
      int c = tid + j * 256, row = c >> 2, col = (c & 3) * 8;
      async_cp16(A + (long long)(m0 + row) * ldA + kt + col,
                 &As[buf][row * LDS_STRIDE + col]);
    }
#pragma unroll
    for (int j = 0; j < 2; ++j) {                 // B: 512 chunks
      int c = tid + j * 256, row = c >> 2, col = (c & 3) * 8;
      async_cp16(B + (long long)(n0 + row) * ldB + kt + col,
                 &Bs[buf][row * LDS_STRIDE + col]);
    }
  };

  int buf = 0;
  stage(0, 0);

  for (int kt = 0; kt < Kd; kt += 32) {
    wait_async0();            // my async copies for `buf` complete
    __syncthreads();          // everyone's complete; prior compute on buf^1 done
    if (kt + 32 < Kd) stage(buf ^ 1, kt + 32);

    // Load ALL fragments first (16x ds_load_b128), then fire 16 WMMAs
    // back-to-back so the matrix pipe never drains mid-burst.
    // A frag (16x32): lane holds row r, halves = K[g*8 .. +8) and K[16+g*8 .. +8)
    v16bf af[4];
#pragma unroll
    for (int mi = 0; mi < 4; ++mi) {
      const __bf16* p = &As[buf][(wm * 64 + mi * 16 + r) * LDS_STRIDE];
      BFPair t;
      t.lo = *(const v8bf*)(p + g * 8);
      t.hi = *(const v8bf*)(p + 16 + g * 8);
      af[mi] = __builtin_bit_cast(v16bf, t);
    }
    // B frag (32x16): lane holds col r, halves = K[g*8 .. +8) and K[16+g*8 .. +8)
    v16bf bfr[4];
#pragma unroll
    for (int ni = 0; ni < 4; ++ni) {
      const __bf16* p = &Bs[buf][(wn * 64 + ni * 16 + r) * LDS_STRIDE];
      BFPair t;
      t.lo = *(const v8bf*)(p + g * 8);
      t.hi = *(const v8bf*)(p + 16 + g * 8);
      bfr[ni] = __builtin_bit_cast(v16bf, t);
    }
#pragma unroll
    for (int ni = 0; ni < 4; ++ni)
#pragma unroll
      for (int mi = 0; mi < 4; ++mi)
        acc[mi][ni] = __builtin_amdgcn_wmma_f32_16x16x32_bf16(
            false, af[mi], false, bfr[ni], (short)0, acc[mi][ni], false, false);
    buf ^= 1;
  }

  // Epilogue. C/D layout: VGPR e holds row m = e + 8*g, col n = lane&15.
#pragma unroll
  for (int mi = 0; mi < 4; ++mi) {
#pragma unroll
    for (int ni = 0; ni < 4; ++ni) {
      v8f c = acc[mi][ni];
      int mBase = m0 + wm * 64 + mi * 16 + g * 8;
      int n     = n0 + wn * 64 + ni * 16 + r;
      if constexpr (EPI == 0) {
        __bf16* C = (__bf16*)Cout + (long long)bz * offC;
#pragma unroll
        for (int e = 0; e < 8; ++e)
          C[(long long)(mBase + e) * ldC + n] = f2bf(c[e] * scale);
      } else if constexpr (EPI == 1) {
        __bf16* C = (__bf16*)Cout + (long long)bz * offC;
        v8bf o;
#pragma unroll
        for (int e = 0; e < 8; ++e) o[e] = f2bf(c[e] * scale);
        *(v8bf*)(C + (long long)n * ldC + mBase) = o;   // 16B contiguous store
      } else {
        float* C = (float*)Cout + (long long)bz * offC;
#pragma unroll
        for (int e = 0; e < 8; ++e)
          C[(long long)(mBase + e) * ldC + n] = c[e] * scale;
      }
    }
  }
}

// fp32 -> bf16 bulk convert (4 elems/thread)
__launch_bounds__(256)
__global__ void cvt_f32_bf16_k(const float* __restrict__ in,
                               __bf16* __restrict__ out) {
  long long i = ((long long)blockIdx.x * 256 + threadIdx.x) * 4;
  const float4 f = *(const float4*)(in + i);
  v4bf o;
  o[0] = f2bf(f.x); o[1] = f2bf(f.y); o[2] = f2bf(f.z); o[3] = f2bf(f.w);
  *(v4bf*)(out + i) = o;
}

// W[D][U] fp32 -> Wt[U][D] bf16, tiled transpose, 3 weights via blockIdx.z
__launch_bounds__(256)
__global__ void transcvt_w_k(const float* __restrict__ W0,
                             const float* __restrict__ W1,
                             const float* __restrict__ W2,
                             __bf16* __restrict__ Wt) {
  __shared__ float t[32][33];
  const float* W = (blockIdx.z == 0) ? W0 : (blockIdx.z == 1) ? W1 : W2;
  __bf16* O = Wt + (long long)blockIdx.z * DDIM * UDIM;
  int tx = threadIdx.x, ty = threadIdx.y;      // (32, 8)
  int d0 = blockIdx.x * 32, u0 = blockIdx.y * 32;
#pragma unroll
  for (int j = 0; j < 4; ++j)
    t[ty + j * 8][tx] = W[(long long)(d0 + ty + j * 8) * UDIM + u0 + tx];
  __syncthreads();
#pragma unroll
  for (int j = 0; j < 4; ++j)
    O[(long long)(u0 + ty + j * 8) * DDIM + d0 + tx] = f2bf(t[tx][ty + j * 8]);
}

// One block per row: softmax over 2048 fp32 scores -> bf16 probs
__launch_bounds__(256)
__global__ void softmax_rows_k(const float* __restrict__ S,
                               __bf16* __restrict__ P) {
  __shared__ float red[8];
  const long long row = blockIdx.x;
  const float* x = S + row * SDIM;
  const int tid = threadIdx.x;

  float v[8];
  float mx = -3.0e38f;
#pragma unroll
  for (int j = 0; j < 8; ++j) { v[j] = x[tid + j * 256]; mx = fmaxf(mx, v[j]); }
#pragma unroll
  for (int off = 16; off >= 1; off >>= 1) mx = fmaxf(mx, __shfl_xor(mx, off, 32));
  if ((tid & 31) == 0) red[tid >> 5] = mx;
  __syncthreads();
  mx = red[0];
#pragma unroll
  for (int w = 1; w < 8; ++w) mx = fmaxf(mx, red[w]);

  float e[8];
  float s = 0.f;
#pragma unroll
  for (int j = 0; j < 8; ++j) {
    e[j] = exp2f((v[j] - mx) * 1.4426950408889634f);
    s += e[j];
  }
#pragma unroll
  for (int off = 16; off >= 1; off >>= 1) s += __shfl_xor(s, off, 32);
  __syncthreads();
  if ((tid & 31) == 0) red[tid >> 5] = s;
  __syncthreads();
  s = 0.f;
#pragma unroll
  for (int w = 0; w < 8; ++w) s += red[w];
  float inv = 1.0f / s;

  __bf16* p = P + row * SDIM;
#pragma unroll
  for (int j = 0; j < 8; ++j) p[tid + j * 256] = f2bf(e[j] * inv);
}

extern "C" void kernel_launch(void* const* d_in, const int* /*in_sizes*/, int /*n_in*/,
                              void* d_out, int /*out_size*/, void* d_ws, size_t /*ws_size*/,
                              hipStream_t stream) {
  const float* X  = (const float*)d_in[0];
  const float* Wq = (const float*)d_in[1];
  const float* Wk = (const float*)d_in[2];
  const float* Wv = (const float*)d_in[3];
  float* out = (float*)d_out;

  const long long BS = MROWS;           // 16384
  const long long WN = (long long)DDIM * UDIM;

  // Workspace layout (bf16 elems / f32 elems):
  __bf16* Xb = (__bf16*)d_ws;           // [BS][D]           32 MB
  __bf16* Wt = Xb + BS * DDIM;          // 3x [U][D]          6 MB
  __bf16* Qb = Wt + 3 * WN;             // [BS][U]           32 MB
  __bf16* Kb = Qb + BS * UDIM;          // [BS][U]           32 MB
  __bf16* Vt = Kb + BS * UDIM;          // [U][BS] (V^T)     32 MB
  float*  Sc = (float*)(Vt + BS * UDIM);// [B][S][S] scores 134 MB
  __bf16* Pb = (__bf16*)(Sc + (long long)BDIM * SDIM * SDIM); // probs 67 MB

  // 1) Convert inputs to bf16
  cvt_f32_bf16_k<<<dim3((unsigned)(BS * DDIM / 1024)), 256, 0, stream>>>(X, Xb);
  transcvt_w_k<<<dim3(DDIM / 32, UDIM / 32, 3), dim3(32, 8), 0, stream>>>(Wq, Wk, Wv, Wt);

  // 2) QKV projections: [16384x1024] @ Wt^T -> Q,K row-major bf16; V transposed
  dim3 gQKV(MROWS / 256, UDIM / 128, 1);
  gemm_bf16_wmma<0><<<gQKV, 256, 0, stream>>>(Xb, Wt + 0 * WN, Qb,
      DDIM, DDIM, UDIM, DDIM, 0, 0, 0, 1.0f);
  gemm_bf16_wmma<0><<<gQKV, 256, 0, stream>>>(Xb, Wt + 1 * WN, Kb,
      DDIM, DDIM, UDIM, DDIM, 0, 0, 0, 1.0f);
  gemm_bf16_wmma<1><<<gQKV, 256, 0, stream>>>(Xb, Wt + 2 * WN, Vt,
      DDIM, DDIM, MROWS /*ldC of V^T*/, DDIM, 0, 0, 0, 1.0f);

  // 3) Scores: per batch, Q[b] @ K[b]^T * (1/32) -> fp32
  dim3 gS(SDIM / 256, SDIM / 128, BDIM);
  gemm_bf16_wmma<2><<<gS, 256, 0, stream>>>(Qb, Kb, Sc,
      UDIM, UDIM, SDIM, UDIM,
      (long long)SDIM * UDIM, (long long)SDIM * UDIM, (long long)SDIM * SDIM,
      0.03125f);

  // 4) Row softmax -> bf16 probs
  softmax_rows_k<<<dim3(MROWS), 256, 0, stream>>>(Sc, Pb);

  // 5) Out = P[b] @ V[b]: A = probs [S][S], B = Vt [U][BS] (col offset b*S)
  dim3 gPV(SDIM / 256, UDIM / 128, BDIM);
  gemm_bf16_wmma<2><<<gPV, 256, 0, stream>>>(Pb, Vt, out,
      SDIM, MROWS, UDIM, SDIM,
      (long long)SDIM * SDIM, (long long)SDIM /*column offset into V^T*/,
      (long long)SDIM * UDIM,
      1.0f);
}
